// vector_quantizer_5488968204590
// MI455X (gfx1250) — matrix operand order repaired and checked by reading, not compiled, not run on hardware
//
#include <hip/hip_runtime.h>
#include <hip/hip_bf16.h>

typedef __attribute__((ext_vector_type(16))) __bf16 v16bf;
typedef __attribute__((ext_vector_type(8)))  __bf16 v8bf;
typedef __attribute__((ext_vector_type(8)))  float  v8f;
typedef unsigned short ushort_t;

#define D_DIM 256
#define K_DIM 1024
#define N_ROWS 32768
#define Q_OFF    0ull
#define LOSS_OFF 8388608ull
#define PERP_OFF 8388609ull
#define ENC_OFF  8388610ull
#define IDX_OFF  41943042ull

__device__ __forceinline__ ushort_t f2bf(float f) {
    unsigned u = __float_as_uint(f);
    u += 0x7FFFu + ((u >> 16) & 1u);           // round-to-nearest-even
    return (ushort_t)(u >> 16);
}
__device__ __forceinline__ float bf2f(ushort_t h) {
    return __uint_as_float(((unsigned)h) << 16);
}

// ---------------------------------------------------------------------------
// Kernel 1: build bf16 hi/lo transposed codebook, ||w_k||^2, zero accumulators
// grid = K_DIM blocks, 256 threads (one column per block)
// ---------------------------------------------------------------------------
__global__ __launch_bounds__(256) void vq_prep(const float* __restrict__ w,
                                               ushort_t* __restrict__ whiT,
                                               ushort_t* __restrict__ wloT,
                                               float* __restrict__ wnorm,
                                               float* __restrict__ hist,
                                               float* __restrict__ lossAcc) {
    int k = blockIdx.x;          // code index
    int d = threadIdx.x;         // dim index
    float v = w[(size_t)d * K_DIM + k];
    ushort_t h = f2bf(v);
    float hf = bf2f(h);
    ushort_t l = f2bf(v - hf);
    whiT[(size_t)k * D_DIM + d] = h;
    wloT[(size_t)k * D_DIM + d] = l;

    __shared__ float red[256];
    red[d] = v * v;
    __syncthreads();
    for (int s = 128; s > 0; s >>= 1) {
        if (d < s) red[d] += red[d + s];
        __syncthreads();
    }
    if (d == 0) {
        wnorm[k] = red[0];
        hist[k]  = 0.0f;
        if (k == 0) *lossAcc = 0.0f;
    }
}

// ---------------------------------------------------------------------------
// Kernel 2: distances via bf16x3 WMMA, argmin, all large outputs
// grid = N_ROWS/16 = 2048 blocks, 256 threads (8 wave32)
// ---------------------------------------------------------------------------
__global__ __launch_bounds__(256) void vq_main(const float* __restrict__ x,
                                               const float* __restrict__ w,
                                               const ushort_t* __restrict__ whiT,
                                               const ushort_t* __restrict__ wloT,
                                               const float* __restrict__ wnorm,
                                               float* __restrict__ hist,
                                               float* __restrict__ lossAcc,
                                               float* __restrict__ out) {
    __shared__ alignas(32) ushort_t sAhi[16 * D_DIM];
    __shared__ alignas(32) ushort_t sAlo[16 * D_DIM];
    __shared__ float sX[16 * D_DIM];
    __shared__ float sScore[2048];   // [wave][row][colmod]
    __shared__ int   sIdx[2048];
    __shared__ int   sBest[16];

    const int tid = threadIdx.x;
    const size_t rowBase = (size_t)blockIdx.x * 16;

    // Stage the 16-row input tile: f32 copy + bf16 hi/lo split.
    for (int m = 0; m < 16; ++m) {
        float v = x[(rowBase + m) * D_DIM + tid];
        sX[m * D_DIM + tid] = v;
        ushort_t h = f2bf(v);
        sAhi[m * D_DIM + tid] = h;
        sAlo[m * D_DIM + tid] = f2bf(v - bf2f(h));
    }
    __syncthreads();

    const int lane = tid & 31;
    const int wave = tid >> 5;
    const int half = lane >> 4;      // 0: lanes 0-15, 1: lanes 16-31
    const int lmod = lane & 15;

    float best[8];
    int   bidx[8];
#pragma unroll
    for (int r = 0; r < 8; ++r) { best[r] = 3.0e38f; bidx[r] = 0; }

    // Each wave sweeps col-tiles wave, wave+8, ... (64 tiles of 16 codes).
    for (int t = wave; t < 64; t += 8) {
        v8f acc = {};
        const int n = t * 16 + lmod;                       // this lane's column
        const ushort_t* bh = whiT + (size_t)n * D_DIM;
        const ushort_t* bl = wloT + (size_t)n * D_DIM;
#pragma unroll
        for (int kk = 0; kk < 8; ++kk) {
            // B fragment: 16 contiguous K values per lane.
            const int kbB = kk * 32 + half * 16;
            v16bf bhi = *(const v16bf*)(bh + kbB);
            v16bf blo = *(const v16bf*)(bl + kbB);
            // A fragment: two 8-element chunks (K=kb..kb+7 and kb+8..kb+15).
            const int kbA = kk * 32 + half * 16;
            const ushort_t* ah = sAhi + lmod * D_DIM + kbA;
            const ushort_t* al = sAlo + lmod * D_DIM + kbA;
            v8bf ah0 = *(const v8bf*)(ah);
            v8bf ah1 = *(const v8bf*)(ah + 8);
            v8bf al0 = *(const v8bf*)(al);
            v8bf al1 = *(const v8bf*)(al + 8);
            v16bf ahi = __builtin_shufflevector(ah0, ah1, 0,1,2,3,4,5,6,7,8,9,10,11,12,13,14,15);
            v16bf alo = __builtin_shufflevector(al0, al1, 0,1,2,3,4,5,6,7,8,9,10,11,12,13,14,15);
            // bf16x3: hi*hi + hi*lo + lo*hi, f32 accumulate (~2^-16 accurate)
            acc = __builtin_amdgcn_wmma_f32_16x16x32_bf16(false, ahi, false, bhi, (short)0, acc, false, false);
            acc = __builtin_amdgcn_wmma_f32_16x16x32_bf16(false, ahi, false, blo, (short)0, acc, false, false);
            acc = __builtin_amdgcn_wmma_f32_16x16x32_bf16(false, alo, false, bhi, (short)0, acc, false, false);
        }
        const float wn = wnorm[n];
#pragma unroll
        for (int r = 0; r < 8; ++r) {
            float s = wn - 2.0f * acc[r];   // ||x||^2 constant per row: dropped
            if (s < best[r]) { best[r] = s; bidx[r] = n; }
        }
    }

    // Publish per-wave candidates: slot r -> row (r + 8*half), col lmod.
#pragma unroll
    for (int r = 0; r < 8; ++r) {
        int m = r + half * 8;
        int e = (wave * 16 + m) * 16 + lmod;
        sScore[e] = best[r];
        sIdx[e]   = bidx[r];
    }
    __syncthreads();

    if (tid < 16) {
        float bs = 3.0e38f; int bi = 0x7FFFFFFF;
        for (int wv = 0; wv < 8; ++wv)
            for (int c = 0; c < 16; ++c) {
                int e = (wv * 16 + tid) * 16 + c;
                float s = sScore[e];
                int   i = sIdx[e];
                if (s < bs || (s == bs && i < bi)) { bs = s; bi = i; }
            }
        sBest[tid] = bi;
    }
    __syncthreads();

    // Streaming outputs: quantized_skip, one-hot encodings, loss partial.
    float lsum = 0.0f;
    for (int m = 0; m < 16; ++m) {
        const int idx = sBest[m];
        const size_t row = rowBase + m;
        float q  = w[(size_t)tid * K_DIM + idx];
        float xv = sX[m * D_DIM + tid];
        float diff = q - xv;
        out[Q_OFF + row * D_DIM + tid] = xv + diff;   // matches x + sg(q - x)
        lsum += diff * diff;
        const size_t eb = ENC_OFF + row * (size_t)K_DIM;
#pragma unroll
        for (int j = 0; j < 4; ++j) {
            int col = tid + j * 256;
            out[eb + col] = (col == idx) ? 1.0f : 0.0f;
        }
    }
    if (tid < 16) {
        out[IDX_OFF + rowBase + tid] = (float)sBest[tid];
        atomicAdd(&hist[sBest[tid]], 1.0f);
    }

    // Block loss reduction (reuse sScore; prior reads completed pre-barrier).
    __syncthreads();
    sScore[tid] = lsum;
    __syncthreads();
    for (int s = 128; s > 0; s >>= 1) {
        if (tid < s) sScore[tid] += sScore[tid + s];
        __syncthreads();
    }
    if (tid == 0) atomicAdd(lossAcc, sScore[0]);
}

// ---------------------------------------------------------------------------
// Kernel 3: perplexity from histogram + final loss scalar. 1 block, 1024 thr.
// ---------------------------------------------------------------------------
__global__ __launch_bounds__(1024) void vq_finish(const float* __restrict__ hist,
                                                  const float* __restrict__ lossAcc,
                                                  float* __restrict__ out) {
    __shared__ float red[1024];
    int t = threadIdx.x;
    float p = hist[t] * (1.0f / (float)N_ROWS);
    red[t] = p * logf(p + 1e-10f);
    __syncthreads();
    for (int s = 512; s > 0; s >>= 1) {
        if (t < s) red[t] += red[t + s];
        __syncthreads();
    }
    if (t == 0) {
        out[LOSS_OFF] = 1.25f * (*lossAcc) * (1.0f / 8388608.0f);
        out[PERP_OFF] = expf(-red[0]);
    }
}

extern "C" void kernel_launch(void* const* d_in, const int* in_sizes, int n_in,
                              void* d_out, int out_size, void* d_ws, size_t ws_size,
                              hipStream_t stream) {
    (void)in_sizes; (void)n_in; (void)out_size; (void)ws_size;
    const float* x = (const float*)d_in[0];   // [32,32,32,256] f32
    const float* w = (const float*)d_in[1];   // [256,1024] f32
    float* out = (float*)d_out;

    unsigned char* ws = (unsigned char*)d_ws;
    ushort_t* whiT   = (ushort_t*)(ws);                 // 512 KB
    ushort_t* wloT   = (ushort_t*)(ws + 524288);        // 512 KB
    float*    wnorm  = (float*)   (ws + 1048576);       // 4 KB
    float*    hist   = (float*)   (ws + 1052672);       // 4 KB
    float*    lossAcc= (float*)   (ws + 1056768);       // 4 B

    vq_prep  <<<K_DIM,        256,  0, stream>>>(w, whiT, wloT, wnorm, hist, lossAcc);
    vq_main  <<<N_ROWS / 16,  256,  0, stream>>>(x, w, whiT, wloT, wnorm, hist, lossAcc, out);
    vq_finish<<<1,            1024, 0, stream>>>(hist, lossAcc, out);
}
